// DeformableTransformer_55722905698789
// MI455X (gfx1250) — matrix operand order repaired and checked by reading, not compiled, not run on hardware
//
#include <hip/hip_runtime.h>
#include <hip/hip_bf16.h>

// ---------------------------------------------------------------------------
// Deformable DETR decoder forward for MI455X (gfx1250, wave32, WMMA).
// ---------------------------------------------------------------------------

typedef __attribute__((ext_vector_type(16))) _Float16 v16h;
typedef __attribute__((ext_vector_type(8)))  float    v8f;

#define D_MODEL   256
#define D_FFN     1024
#define N_HEADS   8
#define HEAD_DIM  32
#define N_LEVELS  4
#define N_POINTS  4
#define N_LAYERS  6
#define BATCH     4
#define N_QUERIES 300
#define LEN_IN    21760
#define ROWS_Q    (BATCH * N_QUERIES)     // 1200
#define ROWS_V    (BATCH * LEN_IN)        // 87040

#define BM 128
#define BN 128
#define BK 32

// ---------------------------------------------------------------------------
// WMMA GEMM:  C[M,N] = act(A[M,K] @ W[N,K]^T + bias)
// A, W, C are f32 in global memory; math in f16 WMMA with f32 accumulate.
// Block: 256 threads = 8 waves arranged 4(M) x 2(N); each wave does 32x64
// (2x4 WMMA 16x16 tiles) -> 8 v_wmma per K-step, A fragments reused 4x.
// Out-of-range A rows are CLAMPED (not zeroed): they only contribute to
// output rows >= M which are never stored, so no load divergence is needed.
// ---------------------------------------------------------------------------
__global__ __launch_bounds__(256) void gemm_wmma_kernel(
    const float* __restrict__ A, const float* __restrict__ W,
    const float* __restrict__ bias, float* __restrict__ C,
    int M, int N, int K, int relu)
{
  __shared__ _Float16 As[BM][BK];   // 8 KB
  __shared__ _Float16 Bs[BN][BK];   // 8 KB

  const int tid  = threadIdx.x;
  const int wave = tid >> 5;
  const int lane = tid & 31;
  const int l16  = lane & 15;
  const int half = lane >> 4;

  const int block_m = blockIdx.x * BM;
  const int block_n = blockIdx.y * BN;
  const int wm = (wave >> 1) * 32;   // 0,32,64,96
  const int wn = (wave & 1) * 64;    // 0,64

  v8f acc[2][4] = {};

  for (int k0 = 0; k0 < K; k0 += BK) {
    // ---- phase 1: issue all global loads (pipelined, single wait) ----
    float4 va[4];
#pragma unroll
    for (int i = 0; i < 4; ++i) {
      int chunk = tid + i * 256;          // 1024 chunks of float4 (128x32)
      int r  = chunk >> 3;                // row 0..127
      int c4 = chunk & 7;                 // float4 col 0..7
      int gm = block_m + r;
      gm = gm < M ? gm : (M - 1);         // clamp: no divergence, rows >= M unused
      va[i] = *reinterpret_cast<const float4*>(A + (size_t)gm * K + k0 + c4 * 4);
    }
    float4 vb[4];
#pragma unroll
    for (int i = 0; i < 4; ++i) {
      int chunk = tid + i * 256;          // 1024 chunks (128x32)
      int r  = chunk >> 3;
      int c4 = chunk & 7;
      int gn = block_n + r;               // N % 128 == 0 for all our GEMMs
      vb[i] = *reinterpret_cast<const float4*>(W + (size_t)gn * K + k0 + c4 * 4);
    }
    // prefetch next A tile into caches (global_prefetch_b8)
    if (k0 + BK < K) {
      int gm = block_m + (tid >> 1);
      gm = gm < M ? gm : (M - 1);
      __builtin_prefetch(A + (size_t)gm * K + k0 + BK + (tid & 1) * 16, 0, 1);
    }
    // ---- phase 2: convert f32 -> f16 and store to LDS ----
#pragma unroll
    for (int i = 0; i < 4; ++i) {
      int chunk = tid + i * 256;
      int r  = chunk >> 3;
      int c4 = chunk & 7;
      _Float16* dst = &As[r][c4 * 4];
      dst[0] = (_Float16)va[i].x; dst[1] = (_Float16)va[i].y;
      dst[2] = (_Float16)va[i].z; dst[3] = (_Float16)va[i].w;
    }
#pragma unroll
    for (int i = 0; i < 4; ++i) {
      int chunk = tid + i * 256;
      int r  = chunk >> 3;
      int c4 = chunk & 7;
      _Float16* dst = &Bs[r][c4 * 4];
      dst[0] = (_Float16)vb[i].x; dst[1] = (_Float16)vb[i].y;
      dst[2] = (_Float16)vb[i].z; dst[3] = (_Float16)vb[i].w;
    }
    __syncthreads();

    // ---- build fragments from LDS ----
    // A 16x32 f16 layout: lane holds row m=l16; elements 0..7 -> K=base8+0..7,
    // elements 8..15 -> K=base8+16..23, base8 = half*8.
    union FragA { v16h v; uint4 u[2]; } af[2];
#pragma unroll
    for (int mt = 0; mt < 2; ++mt) {
      const _Float16* s = &As[wm + mt * 16 + l16][half * 8];
      af[mt].u[0] = *reinterpret_cast<const uint4*>(s);
      af[mt].u[1] = *reinterpret_cast<const uint4*>(s + 16);
    }
    // B 32x16 f16 layout: lane holds col n=l16; elements 0..7 -> K=half*8+0..7, 8..15 -> K=half*8+16..23.
    union FragB { v16h v; uint4 u[2]; } bf[4];
#pragma unroll
    for (int nt = 0; nt < 4; ++nt) {
      const _Float16* s = &Bs[wn + nt * 16 + l16][half * 8];
      bf[nt].u[0] = *reinterpret_cast<const uint4*>(s);
      bf[nt].u[1] = *reinterpret_cast<const uint4*>(s + 16);
    }
#pragma unroll
    for (int mt = 0; mt < 2; ++mt)
#pragma unroll
      for (int nt = 0; nt < 4; ++nt)
        acc[mt][nt] = __builtin_amdgcn_wmma_f32_16x16x32_f16(
            false, af[mt].v, false, bf[nt].v, (short)0, acc[mt][nt], false, false);
    __syncthreads();
  }

  // ---- store D: lanes 0-15 -> N=lane, M=vgpr; lanes 16-31 -> N=lane-16, M=8+vgpr
#pragma unroll
  for (int mt = 0; mt < 2; ++mt) {
#pragma unroll
    for (int nt = 0; nt < 4; ++nt) {
      int n = block_n + wn + nt * 16 + l16;
      float bv = bias ? bias[n] : 0.f;
      int mbase = block_m + wm + mt * 16 + half * 8;
#pragma unroll
      for (int r = 0; r < 8; ++r) {
        int m = mbase + r;
        if (m < M) {
          float val = acc[mt][nt][r] + bv;
          if (relu) val = fmaxf(val, 0.f);
          C[(size_t)m * N + n] = val;
        }
      }
    }
  }
}

// ---------------------------------------------------------------------------
// elementwise add: c = a + b
// ---------------------------------------------------------------------------
__global__ __launch_bounds__(256) void add_kernel(
    const float* __restrict__ a, const float* __restrict__ b,
    float* __restrict__ c, int n)
{
  int i = blockIdx.x * 256 + threadIdx.x;
  if (i < n) c[i] = a[i] + b[i];
}

// ---------------------------------------------------------------------------
// residual + layernorm (in place on state): state = LN(state + t)
// one wave per row of 256
// ---------------------------------------------------------------------------
__global__ __launch_bounds__(256) void add_ln_kernel(
    float* __restrict__ state, const float* __restrict__ t,
    const float* __restrict__ g, const float* __restrict__ beta, int rows)
{
  int wave = threadIdx.x >> 5, lane = threadIdx.x & 31;
  int row = blockIdx.x * 8 + wave;
  if (row >= rows) return;
  size_t base = (size_t)row * D_MODEL;
  float x[8];
  float sum = 0.f, sq = 0.f;
#pragma unroll
  for (int j = 0; j < 8; ++j) {
    int c = lane + 32 * j;
    float v = state[base + c] + t[base + c];
    x[j] = v; sum += v; sq += v * v;
  }
#pragma unroll
  for (int off = 16; off > 0; off >>= 1) {
    sum += __shfl_xor(sum, off);
    sq  += __shfl_xor(sq, off);
  }
  float mu = sum * (1.f / 256.f);
  float var = sq * (1.f / 256.f) - mu * mu;
  float rs = rsqrtf(var + 1e-5f);
#pragma unroll
  for (int j = 0; j < 8; ++j) {
    int c = lane + 32 * j;
    state[base + c] = (x[j] - mu) * rs * g[c] + beta[c];
  }
}

// ---------------------------------------------------------------------------
// MHA attention core: one wave (block of 32) per (b, h, q).
// qk: [ROWS_Q, 512]  (cols 0..255 = Q proj, 256..511 = K proj)
// vp: [ROWS_Q, 256], out: [ROWS_Q, 256]
// ---------------------------------------------------------------------------
__global__ __launch_bounds__(32) void mha_attn_kernel(
    const float* __restrict__ qk, const float* __restrict__ vp,
    float* __restrict__ outb)
{
  int bhq = blockIdx.x;
  int q = bhq % N_QUERIES;
  int bh = bhq / N_QUERIES;
  int h = bh & 7, b = bh >> 3;
  int lane = threadIdx.x;

  __shared__ float p[N_QUERIES];

  size_t qrow = (size_t)(b * N_QUERIES + q) * 512;
  float qv = qk[qrow + h * 32 + lane];
  const float scale = 0.17677669529663687f;   // 1/sqrt(32)

  float s[10];
  float mx = -1e30f;
#pragma unroll
  for (int j = 0; j < 10; ++j) {
    int key = j * 32 + lane;
    float sc = -1e30f;
    if (key < N_QUERIES) {
      const float* kr = qk + (size_t)(b * N_QUERIES + key) * 512 + 256 + h * 32;
      float t = 0.f;
#pragma unroll
      for (int d = 0; d < 32; ++d) t += __shfl(qv, d) * kr[d];
      sc = t * scale;
    }
    s[j] = sc;
    mx = fmaxf(mx, sc);
  }
#pragma unroll
  for (int off = 16; off > 0; off >>= 1) mx = fmaxf(mx, __shfl_xor(mx, off));
  float sum = 0.f;
#pragma unroll
  for (int j = 0; j < 10; ++j) {
    int key = j * 32 + lane;
    float e = (key < N_QUERIES) ? __expf(s[j] - mx) : 0.f;
    s[j] = e; sum += e;
  }
#pragma unroll
  for (int off = 16; off > 0; off >>= 1) sum += __shfl_xor(sum, off);
  float inv = 1.f / sum;
#pragma unroll
  for (int j = 0; j < 10; ++j) {
    int key = j * 32 + lane;
    if (key < N_QUERIES) p[key] = s[j] * inv;
  }
  __syncthreads();

  float acc = 0.f;
  for (int k = 0; k < N_QUERIES; ++k)
    acc += p[k] * vp[(size_t)(b * N_QUERIES + k) * D_MODEL + h * 32 + lane];
  outb[(size_t)(b * N_QUERIES + q) * D_MODEL + h * 32 + lane] = acc;
}

// ---------------------------------------------------------------------------
// softmax over groups of 16 (deformable attention weights), in place.
// aw: [ROWS_Q, 128]; block 128 threads = one (b,q)
// ---------------------------------------------------------------------------
__global__ __launch_bounds__(128) void softmax16_kernel(float* __restrict__ aw)
{
  int bq = blockIdx.x, t = threadIdx.x;
  float v = aw[(size_t)bq * 128 + t];
  float m = v;
#pragma unroll
  for (int off = 8; off > 0; off >>= 1) m = fmaxf(m, __shfl_xor(m, off, 16));
  float e = __expf(v - m);
  float sum = e;
#pragma unroll
  for (int off = 8; off > 0; off >>= 1) sum += __shfl_xor(sum, off, 16);
  aw[(size_t)bq * 128 + t] = e / sum;
}

// ---------------------------------------------------------------------------
// deformable bilinear sampling + weighted sum.
// block 256 = 8 waves (one per head), grid ROWS_Q; lane = channel d (0..31).
// value: [B*LEN_IN, 256]; samp: [ROWS_Q,256]; aw: [ROWS_Q,128]
// ---------------------------------------------------------------------------
__global__ __launch_bounds__(256) void deform_sample_kernel(
    const float* __restrict__ value, const float* __restrict__ samp,
    const float* __restrict__ aw, const float* __restrict__ refp,
    const float* __restrict__ vratio, float* __restrict__ outb)
{
  int bq = blockIdx.x;
  int b = bq / N_QUERIES;
  int h = threadIdx.x >> 5, d = threadIdx.x & 31;

  const int sizes[4]  = {128, 64, 32, 16};
  const int starts[4] = {0, 16384, 20480, 21504};

  float rx = refp[(size_t)bq * 2 + 0];
  float ry = refp[(size_t)bq * 2 + 1];

  float acc = 0.f;
#pragma unroll
  for (int lvl = 0; lvl < 4; ++lvl) {
    int Wsz = sizes[lvl], Hsz = sizes[lvl];
    float Wf = (float)Wsz, Hf = (float)Hsz;
    float refx = rx * vratio[b * 8 + lvl * 2 + 0];
    float refy = ry * vratio[b * 8 + lvl * 2 + 1];
    const float* vbase = value + ((size_t)b * LEN_IN + starts[lvl]) * D_MODEL + h * 32 + d;
#pragma unroll
    for (int pp = 0; pp < 4; ++pp) {
      int oi = (h * 4 + lvl) * 4 + pp;   // h*16 + lvl*4 + pp
      float ox = samp[(size_t)bq * 256 + oi * 2];
      float oy = samp[(size_t)bq * 256 + oi * 2 + 1];
      float w  = aw[(size_t)bq * 128 + oi];
      float x = (refx + ox / Wf) * Wf - 0.5f;
      float y = (refy + oy / Hf) * Hf - 0.5f;
      float x0f = floorf(x), y0f = floorf(y);
      float dx = x - x0f, dy = y - y0f;
      int x0 = (int)x0f, y0 = (int)y0f;
      bool xin0 = (x0 >= 0) && (x0 < Wsz);
      bool xin1 = (x0 + 1 >= 0) && (x0 + 1 < Wsz);
      bool yin0 = (y0 >= 0) && (y0 < Hsz);
      bool yin1 = (y0 + 1 >= 0) && (y0 + 1 < Hsz);
      float v00 = 0.f, v01 = 0.f, v10 = 0.f, v11 = 0.f;
      if (xin0 && yin0) v00 = vbase[(size_t)(y0 * Wsz + x0) * D_MODEL];
      if (xin1 && yin0) v01 = vbase[(size_t)(y0 * Wsz + x0 + 1) * D_MODEL];
      if (xin0 && yin1) v10 = vbase[(size_t)((y0 + 1) * Wsz + x0) * D_MODEL];
      if (xin1 && yin1) v11 = vbase[(size_t)((y0 + 1) * Wsz + x0 + 1) * D_MODEL];
      acc += w * (v00 * (1.f - dx) * (1.f - dy) + v01 * dx * (1.f - dy)
                + v10 * (1.f - dx) * dy        + v11 * dx * dy);
    }
  }
  outb[(size_t)bq * D_MODEL + h * 32 + d] = acc;
}

// ---------------------------------------------------------------------------
// host orchestration
// ---------------------------------------------------------------------------
static inline void launch_gemm(const float* A, const float* W, const float* bias,
                               float* C, int M, int N, int K, int relu,
                               hipStream_t s)
{
  dim3 grid((M + BM - 1) / BM, (N + BN - 1) / BN);
  gemm_wmma_kernel<<<grid, 256, 0, s>>>(A, W, bias, C, M, N, K, relu);
}

extern "C" void kernel_launch(void* const* d_in, const int* in_sizes, int n_in,
                              void* d_out, int out_size, void* d_ws, size_t ws_size,
                              hipStream_t stream)
{
  (void)in_sizes; (void)n_in; (void)out_size; (void)ws_size;

  const float* tgt    = (const float*)d_in[0];
  const float* qpos   = (const float*)d_in[1];
  const float* refpts = (const float*)d_in[2];
  const float* src    = (const float*)d_in[3];
  // d_in[4], d_in[5]: spatial shapes / level starts -- compile-time constants
  const float* vratio = (const float*)d_in[6];

  // params: 7 + layer*22 + idx
  // 0 in_proj_w, 1 in_proj_b, 2 out_proj_w, 3 out_proj_b, 4 samp_w, 5 samp_b,
  // 6 attn_w, 7 attn_b, 8 value_w, 9 value_b, 10 outp_w, 11 outp_b,
  // 12 lin1_w, 13 lin1_b, 14 lin2_w, 15 lin2_b,
  // 16 norm1_g, 17 norm1_b, 18 norm2_g, 19 norm2_b, 20 norm3_g, 21 norm3_b

  float* ws      = (float*)d_ws;
  float* value   = ws;                           // 87040*256
  float* state   = value   + (size_t)ROWS_V * D_MODEL;
  float* qin     = state   + (size_t)ROWS_Q * D_MODEL;
  float* qk      = qin     + (size_t)ROWS_Q * D_MODEL;   // [ROWS_Q,512]
  float* vp      = qk      + (size_t)ROWS_Q * 512;
  float* t2      = vp      + (size_t)ROWS_Q * D_MODEL;
  float* t3      = t2      + (size_t)ROWS_Q * D_MODEL;
  float* sampv   = t3      + (size_t)ROWS_Q * D_MODEL;   // [ROWS_Q,256]
  float* attw    = sampv   + (size_t)ROWS_Q * D_MODEL;   // [ROWS_Q,128]
  float* sampled = attw    + (size_t)ROWS_Q * 128;
  float* ffnh    = sampled + (size_t)ROWS_Q * D_MODEL;   // [ROWS_Q,1024]

  const int nQ = ROWS_Q * D_MODEL;

  hipMemcpyAsync(state, tgt, (size_t)nQ * sizeof(float),
                 hipMemcpyDeviceToDevice, stream);

  for (int layer = 0; layer < N_LAYERS; ++layer) {
    const float* const* P = (const float* const*)(d_in + 7 + layer * 22);
    const float* in_proj_w = P[0];  const float* in_proj_b = P[1];
    const float* out_proj_w = P[2]; const float* out_proj_b = P[3];
    const float* samp_w = P[4];     const float* samp_b = P[5];
    const float* attn_w = P[6];     const float* attn_b = P[7];
    const float* value_w = P[8];    const float* value_b = P[9];
    const float* outp_w = P[10];    const float* outp_b = P[11];
    const float* lin1_w = P[12];    const float* lin1_b = P[13];
    const float* lin2_w = P[14];    const float* lin2_b = P[15];
    const float* n1g = P[16]; const float* n1b = P[17];
    const float* n2g = P[18]; const float* n2b = P[19];
    const float* n3g = P[20]; const float* n3b = P[21];

    // value projection of src (the big GEMM): value = src @ value_w^T + b
    launch_gemm(src, value_w, value_b, value, ROWS_V, D_MODEL, D_MODEL, 0, stream);

    // ---- self attention ----
    add_kernel<<<(nQ + 255) / 256, 256, 0, stream>>>(state, qpos, qin, nQ);
    launch_gemm(qin, in_proj_w, in_proj_b, qk, ROWS_Q, 512, D_MODEL, 0, stream);
    launch_gemm(state, in_proj_w + 512 * D_MODEL, in_proj_b + 512, vp,
                ROWS_Q, D_MODEL, D_MODEL, 0, stream);
    mha_attn_kernel<<<BATCH * N_HEADS * N_QUERIES, 32, 0, stream>>>(qk, vp, t2);
    launch_gemm(t2, out_proj_w, out_proj_b, t3, ROWS_Q, D_MODEL, D_MODEL, 0, stream);
    add_ln_kernel<<<(ROWS_Q + 7) / 8, 256, 0, stream>>>(state, t3, n2g, n2b, ROWS_Q);

    // ---- deformable cross attention ----
    add_kernel<<<(nQ + 255) / 256, 256, 0, stream>>>(state, qpos, qin, nQ);
    launch_gemm(qin, samp_w, samp_b, sampv, ROWS_Q, 256, D_MODEL, 0, stream);
    launch_gemm(qin, attn_w, attn_b, attw, ROWS_Q, 128, D_MODEL, 0, stream);
    softmax16_kernel<<<ROWS_Q, 128, 0, stream>>>(attw);
    deform_sample_kernel<<<ROWS_Q, 256, 0, stream>>>(value, sampv, attw,
                                                     refpts, vratio, sampled);
    launch_gemm(sampled, outp_w, outp_b, t3, ROWS_Q, D_MODEL, D_MODEL, 0, stream);
    add_ln_kernel<<<(ROWS_Q + 7) / 8, 256, 0, stream>>>(state, t3, n1g, n1b, ROWS_Q);

    // ---- FFN ----
    launch_gemm(state, lin1_w, lin1_b, ffnh, ROWS_Q, D_FFN, D_MODEL, 1, stream);
    launch_gemm(ffnh, lin2_w, lin2_b, t3, ROWS_Q, D_MODEL, D_FFN, 0, stream);
    add_ln_kernel<<<(ROWS_Q + 7) / 8, 256, 0, stream>>>(state, t3, n3g, n3b, ROWS_Q);
  }

  hipMemcpyAsync(d_out, state, (size_t)nQ * sizeof(float),
                 hipMemcpyDeviceToDevice, stream);
}
